// Attention_41429254537321
// MI455X (gfx1250) — compile-verified, matrix operand order checked by reading, not run on hardware
//
#include <hip/hip_runtime.h>
#include <hip/hip_bf16.h>

// ---------------------------------------------------------------------------
// Causal MHA (b=4, s=2048, d=1024, h=16, dh=64) for gfx1250 (MI455X).
// All GEMM work (QKV proj, QK^T, PV, out proj) runs on v_wmma_f32_16x16x32_bf16
// with f32 accumulation; softmax in f32 VALU. ~103 GFLOP total -> matrix-core
// bound, so bf16 WMMA is the right precision/path on this chip.
// Dense GEMMs stage the shared weight tile via the Tensor Data Mover
// (tensor_load_to_lds, double-buffered, s_wait_tensorcnt) so B-operand reads
// come from LDS (ds_load_b128) instead of per-wave global fetches.
// Workspace: ~93 MB of bf16 staging (x, Q, K, V^T, z, W_qkv, W_o).
// ---------------------------------------------------------------------------

typedef unsigned short u16;
typedef __attribute__((ext_vector_type(16))) __bf16 bf16x16;
typedef __attribute__((ext_vector_type(8)))  __bf16 bf16x8;
typedef __attribute__((ext_vector_type(8)))  float  f32x8;
typedef __attribute__((ext_vector_type(4)))  unsigned int u32x4;
typedef __attribute__((ext_vector_type(8)))  int i32x8;
typedef __attribute__((ext_vector_type(4)))  int i32x4;

#define D_MODEL 1024
#define SEQ     2048
#define BATCH   4
#define NHEADS  16
#define DHEAD   64
#define NTOK    (BATCH * SEQ)        // 8192
#define NQKV    (3 * D_MODEL)        // 3072

#if __has_builtin(__builtin_amdgcn_tensor_load_to_lds) && \
    __has_builtin(__builtin_amdgcn_s_wait_tensorcnt)
#define USE_TDM 1
#else
#define USE_TDM 0
#endif

// round-to-nearest-even f32 -> bf16 (as raw u16)
__device__ __forceinline__ u16 f2bf(float x) {
  unsigned u = __float_as_uint(x);
  unsigned r = u + 0x7FFFu + ((u >> 16) & 1u);
  return (u16)(r >> 16);
}

// Load one 16-bit A/B WMMA fragment. `p` must already point at this lane's
// row base (row = lane&15) plus this lane-half's +8-element shift. Per-lane
// data is two aligned 16-byte chunks at +0 and +16 elements (K 0..7/16..23
// for lanes 0-15, K 8..15/24..31 for lanes 16-31).
__device__ __forceinline__ bf16x16 load_frag(const u16* p) {
  bf16x8 lo = *reinterpret_cast<const bf16x8*>(p);
  bf16x8 hi = *reinterpret_cast<const bf16x8*>(p + 16);
  bf16x16 r;
#pragma unroll
  for (int i = 0; i < 8; ++i) { r[i] = lo[i]; r[i + 8] = hi[i]; }
  return r;
}

__device__ __forceinline__ f32x8 wmma_bf16(bf16x16 a, bf16x16 b, f32x8 c) {
  return __builtin_amdgcn_wmma_f32_16x16x32_bf16(
      /*neg_a=*/false, a, /*neg_b=*/false, b,
      /*c_mod=*/(short)0, c, /*reuse_a=*/false, /*reuse_b=*/false);
}

__device__ __forceinline__ f32x8 zero8() {
  f32x8 z;
#pragma unroll
  for (int i = 0; i < 8; ++i) z[i] = 0.0f;
  return z;
}

#if USE_TDM
// Issue a TDM load of a 2D bf16 tile [tile_d1 rows x tile_d0 elems] from
// global (row stride = stride_d0 elements) into LDS at byte offset lds_off.
// D# group0/group1 bitfields per CDNA5 ISA:
//   g0: count=1 | lds_addr | global_addr[56:0] | type=2
//   g1: data_size=1(2B) | tensor_dim0/1 | tile_dim0/1 | tensor_dim0_stride
__device__ __forceinline__ void tdm_load_tile_bf16(
    unsigned lds_off, const u16* gptr,
    unsigned tensor_d0, unsigned tensor_d1, unsigned stride_d0,
    unsigned tile_d0, unsigned tile_d1) {
  unsigned long long ga = (unsigned long long)(__UINTPTR_TYPE__)gptr;
  u32x4 g0;
  g0[0] = 1u;                                        // count=1, user D#
  g0[1] = lds_off;                                   // lds_addr (bytes)
  g0[2] = (unsigned)(ga & 0xffffffffu);              // global_addr[31:0]
  g0[3] = (unsigned)((ga >> 32) & 0x01ffffffu)       // global_addr[56:32]
        | (2u << 30);                                // type=2 ("image")
  i32x8 g1;
  g1[0] = (int)(1u << 16);                           // data_size=1 -> 2 bytes
  g1[1] = (int)((tensor_d0 & 0xffffu) << 16);        // tensor_dim0[15:0]
  g1[2] = (int)(((tensor_d0 >> 16) & 0xffffu)        // tensor_dim0[31:16]
        | ((tensor_d1 & 0xffffu) << 16));            // tensor_dim1[15:0]
  g1[3] = (int)(((tensor_d1 >> 16) & 0xffffu)        // tensor_dim1[31:16]
        | ((tile_d0 & 0xffffu) << 16));              // tile_dim0
  g1[4] = (int)(tile_d1 & 0xffffu);                  // tile_dim1 (tile_dim2=0)
  g1[5] = (int)stride_d0;                            // tensor_dim0_stride lo32
  g1[6] = 0;
  g1[7] = 0;
  i32x4 gz = {0, 0, 0, 0};
#if __clang_major__ >= 23
  i32x8 gz8 = {0, 0, 0, 0, 0, 0, 0, 0};
  __builtin_amdgcn_tensor_load_to_lds(g0, g1, gz, gz, gz8, 0);
#else
  __builtin_amdgcn_tensor_load_to_lds(g0, g1, gz, gz, 0);
#endif
}
#endif  // USE_TDM

// ---------------------------------------------------------------------------
__global__ void f32_to_bf16_kernel(const float* __restrict__ in,
                                   u16* __restrict__ out, int n) {
  int i = blockIdx.x * blockDim.x + threadIdx.x;
  int stride = gridDim.x * blockDim.x;
  for (; i < n; i += stride) out[i] = f2bf(in[i]);
}

// ---------------------------------------------------------------------------
// QKV projection: C[8192,3072] = X[8192,1024] @ Wqkv^T + b_qkv
// Wave tile: 32(M) x 64(N); block = 4 waves stacked in M sharing one 64-row
// weight tile staged through the TDM (double-buffered).  Epilogue scatters
// bf16 into head-major Q[b,h,s,dh], K[b,h,s,dh], VT[b,h,dh,s].
__global__ __launch_bounds__(128) void qkv_gemm_kernel(
    const u16* __restrict__ Xb, const u16* __restrict__ Wb,
    const float* __restrict__ bias,
    u16* __restrict__ Q, u16* __restrict__ K, u16* __restrict__ VT) {
#if USE_TDM
  __shared__ u16 wtile[2][64 * 32];
#endif
  const int lane = threadIdx.x & 31;
  const int wave = threadIdx.x >> 5;
  const int col  = lane & 15;
  const int mgrp = (lane >> 4) * 8;
  const int eoff = (lane >> 4) * 8;     // +8 element shift for upper lane half

  const int m0 = blockIdx.x * 128 + wave * 32;  // token base
  const int n0 = blockIdx.y * 64;               // qkv-feature base

  f32x8 acc[2][4];
#pragma unroll
  for (int i = 0; i < 2; ++i)
#pragma unroll
    for (int j = 0; j < 4; ++j) acc[i][j] = zero8();

#if USE_TDM
  const u16* wrow0 = Wb + (size_t)n0 * D_MODEL;
  const unsigned lds0 = (unsigned)(__UINTPTR_TYPE__)&wtile[0][0];
  const unsigned lds1 = (unsigned)(__UINTPTR_TYPE__)&wtile[1][0];
  const int nsteps = D_MODEL / 32;
  if (wave == 0)
    tdm_load_tile_bf16(lds0, wrow0, D_MODEL, 64, D_MODEL, 32, 64);
  for (int step = 0; step < nsteps; ++step) {
    const int k = step * 32;
    if (wave == 0) {
      if (step + 1 < nsteps) {
        tdm_load_tile_bf16((step & 1) ? lds0 : lds1, wrow0 + (k + 32),
                           D_MODEL, 64, D_MODEL, 32, 64);
        __builtin_amdgcn_s_wait_tensorcnt(1);  // current tile complete
      } else {
        __builtin_amdgcn_s_wait_tensorcnt(0);
      }
    }
    __syncthreads();
    const u16* wl = &wtile[step & 1][0];
    bf16x16 a0 = load_frag(Xb + (size_t)(m0 + col) * D_MODEL + k + eoff);
    bf16x16 a1 = load_frag(Xb + (size_t)(m0 + 16 + col) * D_MODEL + k + eoff);
#pragma unroll
    for (int j = 0; j < 4; ++j) {
      bf16x16 b = load_frag(wl + (j * 16 + col) * 32 + eoff);  // ds_load_b128
      acc[0][j] = wmma_bf16(a0, b, acc[0][j]);
      acc[1][j] = wmma_bf16(a1, b, acc[1][j]);
    }
    __syncthreads();
  }
#else
  for (int k = 0; k < D_MODEL; k += 32) {
    bf16x16 a0 = load_frag(Xb + (size_t)(m0 + col) * D_MODEL + k + eoff);
    bf16x16 a1 = load_frag(Xb + (size_t)(m0 + 16 + col) * D_MODEL + k + eoff);
#pragma unroll
    for (int j = 0; j < 4; ++j) {
      bf16x16 b = load_frag(Wb + (size_t)(n0 + j * 16 + col) * D_MODEL + k + eoff);
      acc[0][j] = wmma_bf16(a0, b, acc[0][j]);
      acc[1][j] = wmma_bf16(a1, b, acc[1][j]);
    }
  }
#endif

#pragma unroll
  for (int i = 0; i < 2; ++i) {
#pragma unroll
    for (int j = 0; j < 4; ++j) {
      const int f  = n0 + j * 16 + col;
      const float bv = bias[f];
#pragma unroll
      for (int r = 0; r < 8; ++r) {
        const int m  = m0 + i * 16 + mgrp + r;     // global token
        const int bb = m >> 11, ss = m & (SEQ - 1);
        const u16 h16 = f2bf(acc[i][j][r] + bv);
        if (f < D_MODEL) {                                    // Q
          const int hh = f >> 6, c = f & 63;
          Q[(((size_t)(bb * NHEADS + hh)) * SEQ + ss) * DHEAD + c] = h16;
        } else if (f < 2 * D_MODEL) {                         // K
          const int f2 = f - D_MODEL, hh = f2 >> 6, c = f2 & 63;
          K[(((size_t)(bb * NHEADS + hh)) * SEQ + ss) * DHEAD + c] = h16;
        } else {                                              // V -> V^T
          const int f2 = f - 2 * D_MODEL, hh = f2 >> 6, c = f2 & 63;
          VT[(((size_t)(bb * NHEADS + hh)) * DHEAD + c) * SEQ + ss] = h16;
        }
      }
    }
  }
}

// ---------------------------------------------------------------------------
// Flash attention, causal. One wave per 16-row q tile; 32 keys per chunk.
// Per chunk: 4 WMMAs (S=QK^T), online softmax (16-lane shfl reductions),
// P staged through LDS (D-layout -> A-layout), 4 WMMAs (O += P V).
__global__ __launch_bounds__(128) void attention_kernel(
    const u16* __restrict__ Q, const u16* __restrict__ K,
    const u16* __restrict__ VT, u16* __restrict__ Z) {
  __shared__ u16 pbuf[4][16 * 32];

  const int lane = threadIdx.x & 31;
  const int wave = threadIdx.x >> 5;
  const int col  = lane & 15;
  const int mgrp = (lane >> 4) * 8;
  const int eoff = (lane >> 4) * 8;

  const int bh = blockIdx.x;                    // 0..63 = b*16+h
  const int q0 = (blockIdx.y * 4 + wave) * 16;  // q-tile base

  const u16* Qh = Q  + (size_t)bh * SEQ * DHEAD;
  const u16* Kh = K  + (size_t)bh * SEQ * DHEAD;
  const u16* Vh = VT + (size_t)bh * DHEAD * SEQ;

  // Q A-fragments (rows q0+col, dh chunks 0..31 and 32..63)
  const bf16x16 qa0 = load_frag(Qh + (size_t)(q0 + col) * DHEAD + 0  + eoff);
  const bf16x16 qa1 = load_frag(Qh + (size_t)(q0 + col) * DHEAD + 32 + eoff);

  f32x8 o[4];
#pragma unroll
  for (int t = 0; t < 4; ++t) o[t] = zero8();
  float mrow[8], lrow[8];
#pragma unroll
  for (int r = 0; r < 8; ++r) { mrow[r] = -3.0e38f; lrow[r] = 0.0f; }

  const float scale = 0.125f;  // 1/sqrt(64)
  const int kend = q0 + 16;    // causal: keys <= q index

  for (int kb = 0; kb < kend; kb += 32) {
    // prefetch next chunk's K rows (global_prefetch_b8)
    if (kb + 32 < kend) {
      __builtin_prefetch(Kh + (size_t)(kb + 32 + col) * DHEAD, 0, 1);
      __builtin_prefetch(Kh + (size_t)(kb + 48 + col) * DHEAD, 0, 1);
    }

    // --- S = Q K^T for two 16-key subtiles ---
    f32x8 s[2];
#pragma unroll
    for (int t = 0; t < 2; ++t) {
      const int krow = kb + t * 16 + col;
      bf16x16 b0 = load_frag(Kh + (size_t)krow * DHEAD + 0  + eoff);
      bf16x16 b1 = load_frag(Kh + (size_t)krow * DHEAD + 32 + eoff);
      f32x8 a = zero8();
      a = wmma_bf16(qa0, b0, a);
      a = wmma_bf16(qa1, b1, a);
      s[t] = a;
    }

    // --- scale + causal mask (element (m,n): m=mgrp+r, n=col) ---
#pragma unroll
    for (int t = 0; t < 2; ++t)
#pragma unroll
      for (int r = 0; r < 8; ++r) {
        const int ki = kb + t * 16 + col;
        const int qi = q0 + mgrp + r;
        s[t][r] = (ki <= qi) ? s[t][r] * scale : -3.0e38f;
      }

    // --- online softmax per row, reductions over 16-lane groups ---
#pragma unroll
    for (int r = 0; r < 8; ++r) {
      float rmax = fmaxf(s[0][r], s[1][r]);
#pragma unroll
      for (int off = 1; off < 16; off <<= 1)
        rmax = fmaxf(rmax, __shfl_xor(rmax, off, 16));
      const float mnew  = fmaxf(mrow[r], rmax);
      const float alpha = __expf(mrow[r] - mnew);
      const float p0 = __expf(s[0][r] - mnew);
      const float p1 = __expf(s[1][r] - mnew);
      float rsum = p0 + p1;
#pragma unroll
      for (int off = 1; off < 16; off <<= 1)
        rsum += __shfl_xor(rsum, off, 16);
      lrow[r] = lrow[r] * alpha + rsum;
      mrow[r] = mnew;
#pragma unroll
      for (int t = 0; t < 4; ++t) o[t][r] *= alpha;
      pbuf[wave][(mgrp + r) * 32 + col]      = f2bf(p0);
      pbuf[wave][(mgrp + r) * 32 + 16 + col] = f2bf(p1);
    }

    // --- P as A-fragment from LDS (compiler inserts s_wait_dscnt) ---
    const bf16x16 pf = load_frag(&pbuf[wave][col * 32 + eoff]);

    // --- O += P V  (V^T rows are dh, contiguous in key) ---
#pragma unroll
    for (int t = 0; t < 4; ++t) {
      const int vrow = t * 16 + col;  // dh index
      bf16x16 vf = load_frag(Vh + (size_t)vrow * SEQ + kb + eoff);
      o[t] = wmma_bf16(pf, vf, o[t]);
    }
  }

  // --- epilogue: z[b, s, h*64+dh] = O / l ---
  const int bb = bh >> 4, hh = bh & 15;
#pragma unroll
  for (int t = 0; t < 4; ++t)
#pragma unroll
    for (int r = 0; r < 8; ++r) {
      const int m = q0 + mgrp + r;
      const float v = o[t][r] / lrow[r];
      Z[((size_t)(bb * SEQ + m)) * D_MODEL + hh * DHEAD + t * 16 + col] = f2bf(v);
    }
}

// ---------------------------------------------------------------------------
// Output projection: out[8192,1024] = Z[8192,1024] @ Wo^T + b_o (f32 out)
__global__ __launch_bounds__(128) void oproj_gemm_kernel(
    const u16* __restrict__ Zb, const u16* __restrict__ Wb,
    const float* __restrict__ bias, float* __restrict__ out) {
#if USE_TDM
  __shared__ u16 wtile[2][64 * 32];
#endif
  const int lane = threadIdx.x & 31;
  const int wave = threadIdx.x >> 5;
  const int col  = lane & 15;
  const int mgrp = (lane >> 4) * 8;
  const int eoff = (lane >> 4) * 8;

  const int m0 = blockIdx.x * 128 + wave * 32;
  const int n0 = blockIdx.y * 64;

  f32x8 acc[2][4];
#pragma unroll
  for (int i = 0; i < 2; ++i)
#pragma unroll
    for (int j = 0; j < 4; ++j) acc[i][j] = zero8();

#if USE_TDM
  const u16* wrow0 = Wb + (size_t)n0 * D_MODEL;
  const unsigned lds0 = (unsigned)(__UINTPTR_TYPE__)&wtile[0][0];
  const unsigned lds1 = (unsigned)(__UINTPTR_TYPE__)&wtile[1][0];
  const int nsteps = D_MODEL / 32;
  if (wave == 0)
    tdm_load_tile_bf16(lds0, wrow0, D_MODEL, 64, D_MODEL, 32, 64);
  for (int step = 0; step < nsteps; ++step) {
    const int k = step * 32;
    if (wave == 0) {
      if (step + 1 < nsteps) {
        tdm_load_tile_bf16((step & 1) ? lds0 : lds1, wrow0 + (k + 32),
                           D_MODEL, 64, D_MODEL, 32, 64);
        __builtin_amdgcn_s_wait_tensorcnt(1);
      } else {
        __builtin_amdgcn_s_wait_tensorcnt(0);
      }
    }
    __syncthreads();
    const u16* wl = &wtile[step & 1][0];
    bf16x16 a0 = load_frag(Zb + (size_t)(m0 + col) * D_MODEL + k + eoff);
    bf16x16 a1 = load_frag(Zb + (size_t)(m0 + 16 + col) * D_MODEL + k + eoff);
#pragma unroll
    for (int j = 0; j < 4; ++j) {
      bf16x16 b = load_frag(wl + (j * 16 + col) * 32 + eoff);
      acc[0][j] = wmma_bf16(a0, b, acc[0][j]);
      acc[1][j] = wmma_bf16(a1, b, acc[1][j]);
    }
    __syncthreads();
  }
#else
  for (int k = 0; k < D_MODEL; k += 32) {
    bf16x16 a0 = load_frag(Zb + (size_t)(m0 + col) * D_MODEL + k + eoff);
    bf16x16 a1 = load_frag(Zb + (size_t)(m0 + 16 + col) * D_MODEL + k + eoff);
#pragma unroll
    for (int j = 0; j < 4; ++j) {
      bf16x16 b = load_frag(Wb + (size_t)(n0 + j * 16 + col) * D_MODEL + k + eoff);
      acc[0][j] = wmma_bf16(a0, b, acc[0][j]);
      acc[1][j] = wmma_bf16(a1, b, acc[1][j]);
    }
  }
#endif

#pragma unroll
  for (int i = 0; i < 2; ++i)
#pragma unroll
    for (int j = 0; j < 4; ++j) {
      const int f = n0 + j * 16 + col;
      const float bv = bias[f];
#pragma unroll
      for (int r = 0; r < 8; ++r) {
        const int m = m0 + i * 16 + mgrp + r;
        out[(size_t)m * D_MODEL + f] = acc[i][j][r] + bv;
      }
    }
}

// ---------------------------------------------------------------------------
extern "C" void kernel_launch(void* const* d_in, const int* in_sizes, int n_in,
                              void* d_out, int out_size, void* d_ws, size_t ws_size,
                              hipStream_t stream) {
  const float* x     = (const float*)d_in[0];
  const float* W_qkv = (const float*)d_in[1];
  const float* b_qkv = (const float*)d_in[2];
  const float* W_o   = (const float*)d_in[3];
  const float* b_o   = (const float*)d_in[4];
  float* out = (float*)d_out;

  const size_t N_X  = (size_t)NTOK * D_MODEL;    // 8388608
  const size_t N_WQ = (size_t)NQKV * D_MODEL;    // 3145728
  const size_t N_WO = (size_t)D_MODEL * D_MODEL; // 1048576

  u16* ws  = (u16*)d_ws;
  u16* Xb  = ws;
  u16* Qb  = Xb + N_X;
  u16* Kb  = Qb + N_X;
  u16* Vb  = Kb + N_X;       // stored transposed: [b,h,dh,s]
  u16* Zb  = Vb + N_X;
  u16* Wqb = Zb + N_X;
  u16* Wob = Wqb + N_WQ;     // total ~93 MB of workspace

  f32_to_bf16_kernel<<<512, 256, 0, stream>>>(x, Xb, (int)N_X);
  f32_to_bf16_kernel<<<512, 256, 0, stream>>>(W_qkv, Wqb, (int)N_WQ);
  f32_to_bf16_kernel<<<256, 256, 0, stream>>>(W_o, Wob, (int)N_WO);

  qkv_gemm_kernel<<<dim3(NTOK / 128, NQKV / 64), 128, 0, stream>>>(
      Xb, Wqb, b_qkv, Qb, Kb, Vb);

  attention_kernel<<<dim3(BATCH * NHEADS, (SEQ / 16) / 4), 128, 0, stream>>>(
      Qb, Kb, Vb, Zb);

  oproj_gemm_kernel<<<dim3(NTOK / 128, D_MODEL / 64), 128, 0, stream>>>(
      Zb, Wob, b_o, out);
}
